// SAGEPredictor_10488310137590
// MI455X (gfx1250) — compile-verified, matrix operand order checked by reading, not run on hardware
//
#include <hip/hip_runtime.h>
#include <hip/hip_bf16.h>

typedef _Float16 v16h __attribute__((ext_vector_type(16)));
typedef float    v8f  __attribute__((ext_vector_type(8)));

#define N_NODES  100000
#define N_EDGES  1600000
#define FEAT     128
#define N_GRAPHS 256

// ---------------------------------------------------------------------------
// Utility kernels
// ---------------------------------------------------------------------------
__global__ void zero_kernel(float* __restrict__ p, long long n) {
    long long i = (long long)blockIdx.x * blockDim.x + threadIdx.x;
    long long stride = (long long)gridDim.x * blockDim.x;
    for (; i < n; i += stride) p[i] = 0.0f;
}

// Convert f32 weight [K=128][N=128] into f16 *pre-swizzled* into the exact
// per-lane WMMA B-fragment order, so the GEMM kernel loads each v16h
// fragment as two coalesced global_load_b128:
//   packed[(((wave*4 + kk)*32 + lane)*16 + e] = W[k(kk,hi,e)*128 + 16*wave + (lane&15)]
// with k(kk,hi,e) = kk*32 + (e<8 ? 8*hi+e : 16 + 8*hi + (e-8)),  hi = lane>>4.
__global__ void cvt_pack_kernel(const float* __restrict__ w,
                                _Float16* __restrict__ o) {
    int i = blockIdx.x * blockDim.x + threadIdx.x;   // 0 .. 16383
    if (i >= FEAT * FEAT) return;
    int e    = i & 15;
    int lane = (i >> 4) & 31;
    int kk   = (i >> 9) & 3;
    int wv   = (i >> 11) & 7;
    int hi   = lane >> 4;
    int k = kk * 32 + ((e < 8) ? (8 * hi + e) : (16 + 8 * hi + (e - 8)));
    int n = 16 * wv + (lane & 15);
    o[i] = (_Float16)w[k * FEAT + n];
}

__global__ void index_count_kernel(const int* __restrict__ idx,
                                   float* __restrict__ acc, int n) {
    int i = blockIdx.x * blockDim.x + threadIdx.x;
    if (i < n) atomicAdd(&acc[idx[i]], 1.0f);
}

// ---------------------------------------------------------------------------
// Edge scatter: agg[dst] += feat[src].  One edge per 32-lane slice,
// 4 coalesced float atomics per lane (128 floats per edge).
// ---------------------------------------------------------------------------
__global__ __launch_bounds__(256)
void scatter_kernel(const float* __restrict__ feat,
                    const int* __restrict__ src,
                    const int* __restrict__ dst,
                    float* __restrict__ agg, int nEdges) {
    long long t = (long long)blockIdx.x * 256 + threadIdx.x;
    int e    = (int)(t >> 5);
    int lane = (int)(t & 31);
    if (e < nEdges) {
        int s = src[e];
        int d = dst[e];
        const float* fr = feat + (size_t)s * FEAT;
        float*       to = agg  + (size_t)d * FEAT;
#pragma unroll
        for (int j = 0; j < 4; ++j)
            atomicAdd(&to[lane + 32 * j], fr[lane + 32 * j]);
    }
}

// ---------------------------------------------------------------------------
// Fused SAGE layer: out = relu( (agg/deg) @ Wl + bias + hin @ Wr )
// Block = 256 threads = 8 waves; block covers 16 node rows (grid 6250 exact),
// wave w covers columns [16w,16w+16). K=128 in 4 chunks of 32, two
// v_wmma_f32_16x16x32_f16 per chunk into one f32 accumulator.
// Weights arrive pre-swizzled (see cvt_pack_kernel): fragment load is a
// single 32B vector load per matrix per chunk.
// ---------------------------------------------------------------------------
__device__ __forceinline__ int frag_k(int k0, int hi, int e) {
    return k0 + ((e < 8) ? (8 * hi + e) : (16 + 8 * hi + (e - 8)));
}

__global__ __launch_bounds__(256)
void sage_gemm_kernel(const float* __restrict__ agg,
                      const float* __restrict__ deg,
                      const float* __restrict__ hin,
                      const _Float16* __restrict__ Wl,   // packed fragments
                      const _Float16* __restrict__ Wr,   // packed fragments
                      const float* __restrict__ bias,    // [128]
                      float* __restrict__ hout,          // N x 128 or null
                      float* __restrict__ pooled,        // 256 x 128 or null
                      const int* __restrict__ batch) {
    const int wave     = threadIdx.x >> 5;     // 0..7 -> column tile
    const int lane     = threadIdx.x & 31;
    const int hi       = lane >> 4;
    const int m        = lane & 15;
    const int row_base = blockIdx.x * 16;
    const int node     = row_base + m;

    const float invdeg = 1.0f / fmaxf(deg[node], 1.0f);

    const float* aggRow = agg + (size_t)node * FEAT;
    const float* xRow   = hin + (size_t)node * FEAT;

    v8f c = {};
#pragma unroll
    for (int kk = 0; kk < 4; ++kk) {
        // B fragments: contiguous 32B per lane, coalesced across the wave.
        const size_t fb = ((size_t)(wave * 4 + kk) * 32 + lane) * 16;
        v16h bl = *(const v16h*)(Wl + fb);
        v16h br = *(const v16h*)(Wr + fb);

        v16h a1, a2;
#pragma unroll
        for (int e = 0; e < 16; ++e) {
            int k = frag_k(kk * 32, hi, e);
            a1[e] = (_Float16)(aggRow[k] * invdeg);
            a2[e] = (_Float16)(xRow[k]);
        }
        c = __builtin_amdgcn_wmma_f32_16x16x32_f16(
                false, a1, false, bl, (short)0, c, false, false);
        c = __builtin_amdgcn_wmma_f32_16x16x32_f16(
                false, a2, false, br, (short)0, c, false, false);
    }

    // C layout: VGPR r -> M = r + 8*hi, N = lane&15 (+ 16*wave).
    const int outcol = wave * 16 + (lane & 15);
    const float b    = bias[outcol];
    if (hout != nullptr) {
#pragma unroll
        for (int r = 0; r < 8; ++r) {
            int onode = row_base + r + 8 * hi;
            hout[(size_t)onode * FEAT + outcol] = fmaxf(c[r] + b, 0.0f);
        }
    } else {
#pragma unroll
        for (int r = 0; r < 8; ++r) {
            int onode = row_base + r + 8 * hi;
            float v = fmaxf(c[r] + b, 0.0f);
            atomicAdd(&pooled[(size_t)batch[onode] * FEAT + outcol], v);
        }
    }
}

// ---------------------------------------------------------------------------
// Head: out[g] = dot(pooled[g]/cnt[g], Wfc) + bfc.  One wave per graph.
// ---------------------------------------------------------------------------
__global__ __launch_bounds__(32)
void head_kernel(const float* __restrict__ pooled,
                 const float* __restrict__ cnt,
                 const float* __restrict__ Wfc,
                 const float* __restrict__ bfc,
                 float* __restrict__ out) {
    int g = blockIdx.x;
    int lane = threadIdx.x;
    float s = 0.0f;
    for (int j = lane; j < FEAT; j += 32)
        s += pooled[(size_t)g * FEAT + j] * Wfc[j];
#pragma unroll
    for (int off = 16; off > 0; off >>= 1)
        s += __shfl_down(s, off, 32);
    if (lane == 0)
        out[g] = s / fmaxf(cnt[g], 1.0f) + bfc[0];
}

// ---------------------------------------------------------------------------
extern "C" void kernel_launch(void* const* d_in, const int* in_sizes, int n_in,
                              void* d_out, int out_size, void* d_ws, size_t ws_size,
                              hipStream_t stream) {
    const float* x     = (const float*)d_in[0];
    const int*   ei    = (const int*)d_in[1];
    const int*   batch = (const int*)d_in[2];
    const float* W1l   = (const float*)d_in[3];
    const float* b1    = (const float*)d_in[4];
    const float* W1r   = (const float*)d_in[5];
    const float* W2l   = (const float*)d_in[6];
    const float* b2    = (const float*)d_in[7];
    const float* W2r   = (const float*)d_in[8];
    const float* Wfc   = (const float*)d_in[9];
    const float* bfc   = (const float*)d_in[10];
    const int* src = ei;
    const int* dst = ei + N_EDGES;

    // Workspace carve-up (256B aligned).
    char* ws = (char*)d_ws;
    size_t off = 0;
    auto carve = [&](size_t bytes) -> void* {
        void* p = ws + off;
        off += (bytes + 255) & ~(size_t)255;
        return p;
    };
    float*    deg    = (float*)carve((size_t)N_NODES * 4);
    float*    agg    = (float*)carve((size_t)N_NODES * FEAT * 4);
    float*    h1     = (float*)carve((size_t)N_NODES * FEAT * 4);
    float*    pooled = (float*)carve((size_t)N_GRAPHS * FEAT * 4);
    float*    cnt    = (float*)carve((size_t)N_GRAPHS * 4);
    _Float16* w1l    = (_Float16*)carve((size_t)FEAT * FEAT * 2);
    _Float16* w1r    = (_Float16*)carve((size_t)FEAT * FEAT * 2);
    _Float16* w2l    = (_Float16*)carve((size_t)FEAT * FEAT * 2);
    _Float16* w2r    = (_Float16*)carve((size_t)FEAT * FEAT * 2);

    const int ZB = 2048;           // blocks for grid-stride zeroing
    const long long aggN = (long long)N_NODES * FEAT;

    // 1) Zero accumulators (every call: atomics below accumulate).
    zero_kernel<<<ZB, 256, 0, stream>>>(deg, N_NODES);
    zero_kernel<<<ZB, 256, 0, stream>>>(agg, aggN);
    zero_kernel<<<8, 256, 0, stream>>>(pooled, (long long)N_GRAPHS * FEAT);
    zero_kernel<<<1, 256, 0, stream>>>(cnt, N_GRAPHS);

    // 2) Convert + fragment-swizzle weights to f16 for the WMMA path.
    const int wn = FEAT * FEAT;
    cvt_pack_kernel<<<(wn + 255) / 256, 256, 0, stream>>>(W1l, w1l);
    cvt_pack_kernel<<<(wn + 255) / 256, 256, 0, stream>>>(W1r, w1r);
    cvt_pack_kernel<<<(wn + 255) / 256, 256, 0, stream>>>(W2l, w2l);
    cvt_pack_kernel<<<(wn + 255) / 256, 256, 0, stream>>>(W2r, w2r);

    // 3) In-degree and per-graph node counts.
    index_count_kernel<<<(N_EDGES + 255) / 256, 256, 0, stream>>>(dst, deg, N_EDGES);
    index_count_kernel<<<(N_NODES + 255) / 256, 256, 0, stream>>>(batch, cnt, N_NODES);

    const int scatterGrid = (int)(((long long)N_EDGES * 32 + 255) / 256);
    const int gemmGrid    = N_NODES / 16;   // 6250, exact

    // 4) Layer 1: aggregate + fused dual-GEMM + bias + relu -> h1.
    scatter_kernel<<<scatterGrid, 256, 0, stream>>>(x, src, dst, agg, N_EDGES);
    sage_gemm_kernel<<<gemmGrid, 256, 0, stream>>>(agg, deg, x, w1l, w1r, b1,
                                                   h1, nullptr, nullptr);

    // 5) Layer 2: re-zero agg, aggregate h1, GEMM with fused mean-pool.
    zero_kernel<<<ZB, 256, 0, stream>>>(agg, aggN);
    scatter_kernel<<<scatterGrid, 256, 0, stream>>>(h1, src, dst, agg, N_EDGES);
    sage_gemm_kernel<<<gemmGrid, 256, 0, stream>>>(agg, deg, h1, w2l, w2r, b2,
                                                   nullptr, pooled, batch);

    // 6) Head: per-graph dot with Wfc.
    head_kernel<<<N_GRAPHS, 32, 0, stream>>>(pooled, cnt, Wfc, bfc, (float*)d_out);
}
